// CausalSelfAttention_41807211659752
// MI455X (gfx1250) — matrix operand order repaired and checked by reading, not compiled, not run on hardware
//
#include <hip/hip_runtime.h>
#include <hip/hip_bf16.h>

#define N_EMBD   1024
#define N_HEAD   16
#define HEAD_DIM 64
#define BATCH    4
#define SEQ      2048

typedef __attribute__((ext_vector_type(16))) __bf16 v16bf;
typedef __attribute__((ext_vector_type(8)))  __bf16 v8bf;
typedef __attribute__((ext_vector_type(8)))  float  v8f;
typedef int v4i32 __attribute__((vector_size(16)));

#if __has_builtin(__builtin_amdgcn_global_load_async_to_lds_b128) && \
    __has_builtin(__builtin_amdgcn_s_wait_asynccnt)
#define ASYNC_OK 1
#else
#define ASYNC_OK 0
#endif

// 16-byte global -> LDS copy (per lane). Async (ASYNCcnt path) when the
// toolchain declares the gfx1250 builtin, else plain load+store.
static __device__ __forceinline__ void copy16(const __bf16* g, __bf16* l) {
#if ASYNC_OK
  __builtin_amdgcn_global_load_async_to_lds_b128(
      (__attribute__((address_space(1))) v4i32*)g,
      (__attribute__((address_space(3))) v4i32*)l, 0, 0);
#else
  *(v8bf*)l = *(const v8bf*)g;
#endif
}

template <int N>
static __device__ __forceinline__ void wait_async() {
#if ASYNC_OK
  __builtin_amdgcn_s_wait_asynccnt(N);
#endif
}

// ---- WMMA fragment loaders (bf16 16x16x32, wave32) -------------------------
// A (16x32, MxK): lanes 0-15 row=lane, elems 0-7 = K0..7, 8-15 = K16..23;
// lanes 16-31 row=lane-16, elems 0-7 = K8..15, 8-15 = K24..31.
static __device__ __forceinline__ v16bf load_a_frag(const __bf16* base, int ld) {
  const int lane = threadIdx.x & 31;
  const int row  = lane & 15;
  const int kh   = (lane >> 4) << 3;          // 0 or 8
  const __bf16* p = base + row * ld + kh;
  v8bf lo = *(const v8bf*)(p);
  v8bf hi = *(const v8bf*)(p + 16);
  v16bf a;
#pragma unroll
  for (int i = 0; i < 8; ++i) { a[i] = lo[i]; a[i + 8] = hi[i]; }
  return a;
}

// B (32x16, KxN): lanes 0-15 N=lane, elems 0-7 = K0..7, 8-15 = K16..23;
// lanes 16-31 N=lane-16, elems 0-7 = K8..15, 8-15 = K24..31. Tile stored [N][K].
static __device__ __forceinline__ v16bf load_b_frag(const __bf16* base, int ld) {
  const int lane = threadIdx.x & 31;
  const int n  = lane & 15;
  const int kb = (lane >> 4) << 3;            // 0 or 8
  const __bf16* p = base + n * ld + kb;
  v8bf lo = *(const v8bf*)(p);
  v8bf hi = *(const v8bf*)(p + 16);
  v16bf b;
#pragma unroll
  for (int i = 0; i < 8; ++i) { b[i] = lo[i]; b[i + 8] = hi[i]; }
  return b;
}

static __device__ __forceinline__ v8f wmma_bf16(v16bf a, v16bf b, v8f c) {
  return __builtin_amdgcn_wmma_f32_16x16x32_bf16(false, a, false, b, (short)0, c,
                                                 false, false);
}

// ---- converts ---------------------------------------------------------------
__global__ void cvt_f32_bf16(const float* __restrict__ src,
                             __bf16* __restrict__ dst, int n) {
  int i = blockIdx.x * blockDim.x + threadIdx.x;
  int stride = gridDim.x * blockDim.x;
  for (; i < n; i += stride) dst[i] = (__bf16)src[i];
}

// src [K][N] fp32 -> dst [N][K] bf16 (so GEMM B tiles are straight copies)
__global__ __launch_bounds__(256) void cvt_transpose_bf16(
    const float* __restrict__ src, __bf16* __restrict__ dst, int K, int N) {
  __shared__ float tile[32][33];
  const int k0 = blockIdx.y * 32, n0 = blockIdx.x * 32;
  const int tx = threadIdx.x & 31, ty = threadIdx.x >> 5;   // 8 rows/pass
#pragma unroll
  for (int r = ty; r < 32; r += 8)
    tile[r][tx] = src[(size_t)(k0 + r) * N + n0 + tx];
  __syncthreads();
#pragma unroll
  for (int r = ty; r < 32; r += 8)
    dst[(size_t)(n0 + r) * K + k0 + tx] = (__bf16)tile[tx][r];
}

// ---- shared GEMM mainloop: 128x256 block, 8 waves (2x4), wave tile 64x64 ----
// A [M][K] bf16 row-major; BT [Ntot][K] bf16 row-major (pre-transposed weights)
static __device__ __forceinline__ void gemm_mainloop(
    const __bf16* __restrict__ A, const __bf16* __restrict__ BT, int K,
    int m0, int n0, int wm, int wn, int tid,
    __bf16* lds_a, __bf16* lds_b, v8f acc[4][4]) {
  const int KT = K / 32;

  auto issue_tile = [&](int kt, int buf) {
    __bf16* la = lds_a + buf * (128 * 32);
    __bf16* lb = lds_b + buf * (256 * 32);
    const int k0 = kt * 32;
#pragma unroll
    for (int i = 0; i < 2; ++i) {            // A: 512 16B chunks / 256 thr
      int c = tid + i * 256;
      int row = c >> 2, kc = (c & 3) << 3;
      copy16(A + (size_t)(m0 + row) * K + k0 + kc, la + row * 32 + kc);
    }
#pragma unroll
    for (int i = 0; i < 4; ++i) {            // B: 1024 16B chunks / 256 thr
      int c = tid + i * 256;
      int n = c >> 2, kc = (c & 3) << 3;
      copy16(BT + (size_t)(n0 + n) * K + k0 + kc, lb + n * 32 + kc);
    }
  };

  issue_tile(0, 0);
  for (int kt = 0; kt < KT; ++kt) {
    const int cur = kt & 1;
    if (kt + 1 < KT) {
      issue_tile(kt + 1, cur ^ 1);
      wait_async<6>();                       // previous batch (6 instrs) done
    } else {
      wait_async<0>();
    }
    __syncthreads();

    const __bf16* la = lds_a + cur * (128 * 32);
    const __bf16* lb = lds_b + cur * (256 * 32);
    v16bf af[4], bfr[4];
#pragma unroll
    for (int i = 0; i < 4; ++i)
      af[i] = load_a_frag(la + (wm * 64 + i * 16) * 32, 32);
#pragma unroll
    for (int j = 0; j < 4; ++j)
      bfr[j] = load_b_frag(lb + (wn * 64 + j * 16) * 32, 32);
#pragma unroll
    for (int i = 0; i < 4; ++i)
#pragma unroll
      for (int j = 0; j < 4; ++j) acc[i][j] = wmma_bf16(af[i], bfr[j], acc[i][j]);
    __syncthreads();
  }
}

// ---- QKV GEMM: [8192,1024] x [1024,3072]^T + bias -> q/k [B,H,T,D], v [B,H,D,T]
__global__ __launch_bounds__(256, 1) void qkv_gemm_kernel(
    const __bf16* __restrict__ A, const __bf16* __restrict__ BT,
    const float* __restrict__ bias, __bf16* __restrict__ qd,
    __bf16* __restrict__ kd, __bf16* __restrict__ vd) {
  __shared__ __align__(32) __bf16 lds_a[2 * 128 * 32];
  __shared__ __align__(32) __bf16 lds_b[2 * 256 * 32];
  const int tid = threadIdx.x;
  const int wave = tid >> 5;
  const int wm = wave & 1, wn = wave >> 1;
  const int m0 = blockIdx.x * 128;
  const int n0 = blockIdx.y * 256;

  v8f zero = {};
  v8f acc[4][4];
#pragma unroll
  for (int i = 0; i < 4; ++i)
#pragma unroll
    for (int j = 0; j < 4; ++j) acc[i][j] = zero;

  gemm_mainloop(A, BT, N_EMBD, m0, n0, wm, wn, tid, lds_a, lds_b, acc);

  const int lane = tid & 31;
  const int rbase = (lane >> 4) << 3;
  const int ncol = lane & 15;
#pragma unroll
  for (int i = 0; i < 4; ++i)
#pragma unroll
    for (int j = 0; j < 4; ++j)
#pragma unroll
      for (int r = 0; r < 8; ++r) {
        int row = m0 + wm * 64 + i * 16 + rbase + r;   // b*T + t
        int col = n0 + wn * 64 + j * 16 + ncol;        // 0..3071
        float vv = acc[i][j][r] + bias[col];
        int b = row >> 11, t = row & (SEQ - 1);
        int sel = col >> 10, cc = col & (N_EMBD - 1);
        int h = cc >> 6, d = cc & 63;
        size_t bh = (size_t)(b * N_HEAD + h);
        __bf16 o = (__bf16)vv;
        if (sel == 0)      qd[(bh * SEQ + t) * HEAD_DIM + d] = o;
        else if (sel == 1) kd[(bh * SEQ + t) * HEAD_DIM + d] = o;
        else               vd[(bh * HEAD_DIM + d) * SEQ + t] = o;  // transposed
      }
}

// ---- Flash attention: per (b,h), 64 q-rows per block, 4 waves x 16 rows -----
__global__ __launch_bounds__(128, 1) void flash_attn_kernel(
    const __bf16* __restrict__ q, const __bf16* __restrict__ k,
    const __bf16* __restrict__ vT, __bf16* __restrict__ y) {
  __shared__ __align__(32) __bf16 lds_k[64 * 64];        // [t_key][d]
  __shared__ __align__(32) __bf16 lds_v[64 * 64];        // [d][t_key]
  __shared__ __align__(32) __bf16 lds_p[4 * 16 * 64];    // per-wave P staging

  const int tid = threadIdx.x;
  const int wave = tid >> 5;
  const int lane = tid & 31;
  const int nqt = SEQ / 64;
  const int qt = blockIdx.x % nqt;
  const int bh = blockIdx.x / nqt;
  const int q0 = qt * 64;
  const size_t base  = (size_t)bh * SEQ * HEAD_DIM;      // q,k  [T][D]
  const size_t baseV = (size_t)bh * HEAD_DIM * SEQ;      // vT   [D][T]

  const int arow = lane & 15;
  const int kh = (lane >> 4) << 3;
  const int rbase = (lane >> 4) << 3;
  const int ncol = lane & 15;
  const float scale = 0.125f;                            // 1/sqrt(64)

  // Q fragments (rows q0 + wave*16 .. +15; head dim 64 -> two k-steps)
  v16bf aq[2];
#pragma unroll
  for (int ks = 0; ks < 2; ++ks) {
    const __bf16* p = q + base + (size_t)(q0 + wave * 16 + arow) * 64 + ks * 32 + kh;
    v8bf lo = *(const v8bf*)p;
    v8bf hi = *(const v8bf*)(p + 16);
    v16bf a;
#pragma unroll
    for (int i = 0; i < 8; ++i) { a[i] = lo[i]; a[i + 8] = hi[i]; }
    aq[ks] = a;
  }

  v8f zero = {};
  v8f oacc[4];
#pragma unroll
  for (int j = 0; j < 4; ++j) oacc[j] = zero;
  float mrow[8], lrow[8];
#pragma unroll
  for (int r = 0; r < 8; ++r) { mrow[r] = -1e30f; lrow[r] = 0.0f; }

  for (int kt = 0; kt <= qt; ++kt) {
    __syncthreads();                                     // prev iter done
#pragma unroll
    for (int i = 0; i < 4; ++i) {                        // K tile: straight copy
      int c = tid + i * 128;
      int t = c >> 3, dc = (c & 7) << 3;
      copy16(k + base + (size_t)(kt * 64 + t) * 64 + dc, lds_k + t * 64 + dc);
    }
#pragma unroll
    for (int i = 0; i < 4; ++i) {                        // V tile: straight copy
      int c = tid + i * 128;
      int d = c >> 3, tc = (c & 7) << 3;
      copy16(vT + baseV + (size_t)d * SEQ + kt * 64 + tc, lds_v + d * 64 + tc);
    }
    wait_async<0>();
    __syncthreads();

    // S = Q * K^T  (16 x 64 per wave)
    v8f s[4];
#pragma unroll
    for (int j = 0; j < 4; ++j) {
      v8f c = zero;
      v16bf b0 = load_b_frag(lds_k + (j * 16) * 64 + 0, 64);
      v16bf b1 = load_b_frag(lds_k + (j * 16) * 64 + 32, 64);
      c = wmma_bf16(aq[0], b0, c);
      c = wmma_bf16(aq[1], b1, c);
      s[j] = c;
    }

    // scale + causal mask
#pragma unroll
    for (int j = 0; j < 4; ++j)
#pragma unroll
      for (int r = 0; r < 8; ++r) {
        int grow = q0 + wave * 16 + rbase + r;
        int gcol = kt * 64 + j * 16 + ncol;
        float sv = s[j][r] * scale;
        s[j][r] = (gcol <= grow) ? sv : -1e30f;
      }

    // online softmax (row reductions across 16-lane halves)
    float mnew[8], alpha[8];
#pragma unroll
    for (int r = 0; r < 8; ++r) {
      float mx = fmaxf(fmaxf(s[0][r], s[1][r]), fmaxf(s[2][r], s[3][r]));
#pragma unroll
      for (int m = 1; m < 16; m <<= 1) mx = fmaxf(mx, __shfl_xor(mx, m, 32));
      mnew[r] = fmaxf(mrow[r], mx);
      alpha[r] = __expf(mrow[r] - mnew[r]);
    }
#pragma unroll
    for (int j = 0; j < 4; ++j)
#pragma unroll
      for (int r = 0; r < 8; ++r) s[j][r] = __expf(s[j][r] - mnew[r]);
#pragma unroll
    for (int r = 0; r < 8; ++r) {
      float sum = s[0][r] + s[1][r] + s[2][r] + s[3][r];
#pragma unroll
      for (int m = 1; m < 16; m <<= 1) sum += __shfl_xor(sum, m, 32);
      lrow[r] = lrow[r] * alpha[r] + sum;
      mrow[r] = mnew[r];
    }
#pragma unroll
    for (int j = 0; j < 4; ++j)
#pragma unroll
      for (int r = 0; r < 8; ++r) oacc[j][r] *= alpha[r];

    // stage P (C-layout regs -> row-major bf16 in LDS for A-fragment reload)
    __bf16* pw = lds_p + wave * 16 * 64;
#pragma unroll
    for (int j = 0; j < 4; ++j)
#pragma unroll
      for (int r = 0; r < 8; ++r)
        pw[(rbase + r) * 64 + j * 16 + ncol] = (__bf16)s[j][r];
    __syncthreads();

    // O += P * V
    v16bf ap[2];
#pragma unroll
    for (int ks = 0; ks < 2; ++ks) {
      const __bf16* p = pw + arow * 64 + ks * 32 + kh;
      v8bf lo = *(const v8bf*)p;
      v8bf hi = *(const v8bf*)(p + 16);
      v16bf a;
#pragma unroll
      for (int i = 0; i < 8; ++i) { a[i] = lo[i]; a[i + 8] = hi[i]; }
      ap[ks] = a;
    }
#pragma unroll
    for (int j = 0; j < 4; ++j) {
      v16bf b0 = load_b_frag(lds_v + (j * 16) * 64 + 0, 64);
      v16bf b1 = load_b_frag(lds_v + (j * 16) * 64 + 32, 64);
      oacc[j] = wmma_bf16(ap[0], b0, oacc[j]);
      oacc[j] = wmma_bf16(ap[1], b1, oacc[j]);
    }
  }

  // y[B,T,C] bf16
  const int b = bh >> 4, h = bh & 15;
#pragma unroll
  for (int j = 0; j < 4; ++j)
#pragma unroll
    for (int r = 0; r < 8; ++r) {
      int grow = q0 + wave * 16 + rbase + r;
      float val = oacc[j][r] / lrow[r];
      y[((size_t)(b * SEQ + grow)) * N_EMBD + h * HEAD_DIM + j * 16 + ncol] =
          (__bf16)val;
    }
}

// ---- Output projection: [8192,1024] x [1024,1024]^T + bias -> fp32 ----------
__global__ __launch_bounds__(256, 1) void proj_gemm_kernel(
    const __bf16* __restrict__ A, const __bf16* __restrict__ BT,
    const float* __restrict__ bias, float* __restrict__ out) {
  __shared__ __align__(32) __bf16 lds_a[2 * 128 * 32];
  __shared__ __align__(32) __bf16 lds_b[2 * 256 * 32];
  const int tid = threadIdx.x;
  const int wave = tid >> 5;
  const int wm = wave & 1, wn = wave >> 1;
  const int m0 = blockIdx.x * 128;
  const int n0 = blockIdx.y * 256;

  v8f zero = {};
  v8f acc[4][4];
#pragma unroll
  for (int i = 0; i < 4; ++i)
#pragma unroll
    for (int j = 0; j < 4; ++j) acc[i][j] = zero;

  gemm_mainloop(A, BT, N_EMBD, m0, n0, wm, wn, tid, lds_a, lds_b, acc);

  const int lane = tid & 31;
  const int rbase = (lane >> 4) << 3;
  const int ncol = lane & 15;
#pragma unroll
  for (int i = 0; i < 4; ++i)
#pragma unroll
    for (int j = 0; j < 4; ++j)
#pragma unroll
      for (int r = 0; r < 8; ++r) {
        int row = m0 + wm * 64 + i * 16 + rbase + r;
        int col = n0 + wn * 64 + j * 16 + ncol;
        out[(size_t)row * N_EMBD + col] = acc[i][j][r] + bias[col];
      }
}

extern "C" void kernel_launch(void* const* d_in, const int* in_sizes, int n_in,
                              void* d_out, int out_size, void* d_ws,
                              size_t ws_size, hipStream_t stream) {
  (void)in_sizes; (void)n_in; (void)out_size; (void)ws_size;
  const float* x      = (const float*)d_in[0];
  const float* w_attn = (const float*)d_in[1];
  const float* b_attn = (const float*)d_in[2];
  const float* w_proj = (const float*)d_in[3];
  const float* b_proj = (const float*)d_in[4];
  float* out = (float*)d_out;

  const size_t nx = (size_t)BATCH * SEQ * N_EMBD;      // 8,388,608
  __bf16* ws = (__bf16*)d_ws;
  size_t off = 0;
  __bf16* x_bf      = ws + off; off += nx;
  __bf16* wattnT_bf = ws + off; off += (size_t)N_EMBD * 3 * N_EMBD;  // [3C][C]
  __bf16* wprojT_bf = ws + off; off += (size_t)N_EMBD * N_EMBD;      // [C][C]
  __bf16* q_bf      = ws + off; off += nx;             // [B,H,T,D]
  __bf16* k_bf      = ws + off; off += nx;             // [B,H,T,D]
  __bf16* vT_bf     = ws + off; off += nx;             // [B,H,D,T]
  __bf16* y_bf      = ws + off; off += nx;             // [B,T,C]

  cvt_f32_bf16<<<2048, 256, 0, stream>>>(x, x_bf, (int)nx);
  cvt_transpose_bf16<<<dim3(3 * N_EMBD / 32, N_EMBD / 32), 256, 0, stream>>>(
      w_attn, wattnT_bf, N_EMBD, 3 * N_EMBD);
  cvt_transpose_bf16<<<dim3(N_EMBD / 32, N_EMBD / 32), 256, 0, stream>>>(
      w_proj, wprojT_bf, N_EMBD, N_EMBD);

  qkv_gemm_kernel<<<dim3(64, 12), 256, 0, stream>>>(x_bf, wattnT_bf, b_attn,
                                                    q_bf, k_bf, vT_bf);
  flash_attn_kernel<<<dim3(BATCH * N_HEAD * (SEQ / 64)), 128, 0, stream>>>(
      q_bf, k_bf, vT_bf, y_bf);
  proj_gemm_kernel<<<dim3(64, 4), 256, 0, stream>>>(y_bf, wprojT_bf, b_proj, out);
}
